// MambaLMHeadModel_64321430225085
// MI455X (gfx1250) — matrix operand order repaired and checked by reading, not compiled, not run on hardware
//
#include <hip/hip_runtime.h>
#include <hip/hip_bf16.h>
#include <cstdint>

// ---------------- model constants (from reference) ----------------
constexpr int V_  = 32000;
constexpr int DM_ = 768;
constexpr int NL_ = 4;
constexpr int DI_ = 1536;
constexpr int DTR_= 48;
constexpr int DS_ = 16;
constexpr int DC_ = 4;
constexpr int B_  = 2;
constexpr int L_  = 512;
constexpr int TOK = B_ * L_;          // 1024 rows everywhere
constexpr int XD_ = DTR_ + 2 * DS_;   // 80
constexpr int KDT = 64;               // dt_proj K padded 48 -> 64

typedef __attribute__((ext_vector_type(16))) _Float16 v16h;
typedef __attribute__((ext_vector_type(8)))  _Float16 v8h;
typedef __attribute__((ext_vector_type(8)))  float    v8f;

#if defined(__has_builtin)
#if __has_builtin(__builtin_amdgcn_global_load_async_to_lds_b128)
#define USE_ASYNC_LDS 1
#endif
#endif

#ifdef USE_ASYNC_LDS
// exact pointee type per hipcc diagnostic: 'int __vector(4)' in AS1 / AS3
typedef __attribute__((__vector_size__(4 * sizeof(int)))) int v4i;
using as1_v4i = __attribute__((address_space(1))) v4i*;
using as3_v4i = __attribute__((address_space(3))) v4i*;

__device__ __forceinline__ void wait_async0() {
#if __has_builtin(__builtin_amdgcn_s_wait_asynccnt)
    __builtin_amdgcn_s_wait_asynccnt(0);
#else
    asm volatile("s_wait_asynccnt 0x0" ::: "memory");
#endif
}
#endif

// ---------------- fp32 -> f16 convert with optional zero-pad ----------------
__global__ void convert_pad_f16(const float* __restrict__ src, int srcLd,
                                int rows, int copyCols,
                                _Float16* __restrict__ dst, int dstLd) {
    long long tid = (long long)blockIdx.x * blockDim.x + threadIdx.x;
    long long n = (long long)rows * dstLd;
    if (tid >= n) return;
    int r = (int)(tid / dstLd);
    int c = (int)(tid % dstLd);
    float v = (c < copyCols) ? src[(long long)r * srcLd + c] : 0.f;
    dst[tid] = (_Float16)v;
}

// ---------------- embedding gather ----------------
__global__ void embed_gather(const int* __restrict__ ids,
                             const float* __restrict__ embed,
                             float* __restrict__ x) {
    int tid = blockIdx.x * blockDim.x + threadIdx.x;
    if (tid >= TOK * DM_) return;
    int t = tid / DM_, d = tid % DM_;
    x[tid] = embed[(long long)ids[t] * DM_ + d];
}

// ---------------- RMSNorm -> f16 ----------------
__global__ void rmsnorm_f16(const float* __restrict__ x,
                            const float* __restrict__ w,
                            _Float16* __restrict__ out, int cols) {
    int row = blockIdx.x;
    const float* xr = x + (long long)row * cols;
    float ss = 0.f;
    for (int c = threadIdx.x; c < cols; c += blockDim.x) { float v = xr[c]; ss += v * v; }
    __shared__ float red[256];
    red[threadIdx.x] = ss;
    __syncthreads();
    for (int s = 128; s > 0; s >>= 1) {
        if ((int)threadIdx.x < s) red[threadIdx.x] += red[threadIdx.x + s];
        __syncthreads();
    }
    float scale = rsqrtf(red[0] / cols + 1e-5f);
    for (int c = threadIdx.x; c < cols; c += blockDim.x)
        out[(long long)row * cols + c] = (_Float16)(xr[c] * scale * w[c]);
}

// ---------------- WMMA TN GEMM: C[M,N] = A[M,K] * W[N,K]^T (+C if acc) ----------
// A,W f16 row-major (K-contiguous), C fp32 row-major.
// Block: 256 threads = 8 waves arranged 2(M) x 4(N).
// Wave tile 64x64 = 4x4 of 16x16x32 WMMA -> block tile 128(M) x 256(N).
// K must be a multiple of 32. Out-of-range rows/cols handled by CLAMPED base
// pointers; the corresponding accumulator lanes are never stored.
// B tiles double-buffered through LDS via GLOBAL_LOAD_ASYNC_TO_LDS_B128
// (ASYNCcnt), halving L2 traffic for B.
__global__ void __launch_bounds__(256)
wmma_gemm_tn(const _Float16* __restrict__ A,
             const _Float16* __restrict__ W,
             float* __restrict__ C,
             int M, int N, int K, int acc) {
    const int lane  = threadIdx.x & 31;
    const int wave  = threadIdx.x >> 5;
    const int wm    = blockIdx.y * 128 + (wave >> 2) * 64;
    const int wn    = blockIdx.x * 256 + (wave & 3) * 64;
    const int lr    = lane & 15;                 // row/col within 16-tile
    const int aKsel = (lane & 16) ? 16 : 0;      // A-frag K half select
    const int bKsel = (lane & 16) ? 16 : 0;      // B-frag K half select

    // clamped A row pointers (loop is branch-free)
    const _Float16* aptr[4];
#pragma unroll
    for (int i = 0; i < 4; ++i) {
        int r = wm + 16 * i + lr; if (r > M - 1) r = M - 1;
        aptr[i] = A + (long long)r * K + aKsel;
    }

    v8f c[4][4];
#pragma unroll
    for (int i = 0; i < 4; ++i)
#pragma unroll
        for (int j = 0; j < 4; ++j)
#pragma unroll
            for (int r = 0; r < 8; ++r) c[i][j][r] = 0.f;

#ifdef USE_ASYNC_LDS
    // B tile staged in LDS: 256 rows x 32 halves, row stride 40 halves (80B)
    // -> conflict-free ds_load_b128 fragment reads, 16B aligned.
    constexpr int BSTRIDE = 40;                       // halves
    constexpr unsigned BUFB = 256 * BSTRIDE * 2;      // bytes per buffer
    __shared__ __attribute__((aligned(128))) _Float16 bstage[2][256 * BSTRIDE];

    // each wave issues 4 async b128 per K-step; one instr covers 8 rows x 4 chunks
    const int chunk = lane & 3;                       // 16B chunk within 64B row
    const _Float16* gq[4];
    unsigned lofs[4];
#pragma unroll
    for (int q = 0; q < 4; ++q) {
        int row = wave * 32 + q * 8 + (lane >> 2);    // row within 256-row tile
        int gR = blockIdx.x * 256 + row; if (gR > N - 1) gR = N - 1;
        gq[q] = W + (long long)gR * K + chunk * 8;
        lofs[q] = (unsigned)((row * BSTRIDE + chunk * 8) * 2);
    }
    const unsigned lbase = (unsigned)(uintptr_t)&bstage[0][0];

    auto issueB = [&](int buf, int kt) {
#pragma unroll
        for (int q = 0; q < 4; ++q)
            __builtin_amdgcn_global_load_async_to_lds_b128(
                (as1_v4i)(uintptr_t)(gq[q] + kt),
                (as3_v4i)(uintptr_t)(lbase + (unsigned)buf * BUFB + lofs[q]),
                0, 0);
    };

    issueB(0, 0);
    const int nsteps = K / 32;
    for (int s = 0; s < nsteps; ++s) {
        const int kt = s * 32;
        wait_async0();
        __syncthreads();
        if (s + 1 < nsteps) issueB((s + 1) & 1, kt + 32);

        v16h a[4], b[4];
#pragma unroll
        for (int i = 0; i < 4; ++i) {
            v8h lo = *(const v8h*)(aptr[i] + kt);
            v8h hi = *(const v8h*)(aptr[i] + kt + 8);
#pragma unroll
            for (int t = 0; t < 8; ++t) { a[i][t] = lo[t]; a[i][8 + t] = hi[t]; }
        }
        const _Float16* bbuf = &bstage[s & 1][0];
#pragma unroll
        for (int j = 0; j < 4; ++j)
            b[j] = *(const v16h*)(bbuf + ((wave & 3) * 64 + 16 * j + lr) * BSTRIDE + bKsel);
#pragma unroll
        for (int i = 0; i < 4; ++i)
#pragma unroll
            for (int j = 0; j < 4; ++j)
                c[i][j] = __builtin_amdgcn_wmma_f32_16x16x32_f16(
                    false, a[i], false, b[j], (short)0, c[i][j], false, false);
    }
#else
    // fallback: direct global loads for B (proven path)
    const _Float16* bptr[4];
#pragma unroll
    for (int j = 0; j < 4; ++j) {
        int r = wn + 16 * j + lr; if (r > N - 1) r = N - 1;
        bptr[j] = W + (long long)r * K + bKsel;
    }
    for (int kt = 0; kt < K; kt += 32) {
        v16h a[4], b[4];
#pragma unroll
        for (int i = 0; i < 4; ++i) {
            v8h lo = *(const v8h*)(aptr[i] + kt);
            v8h hi = *(const v8h*)(aptr[i] + kt + 8);
#pragma unroll
            for (int t = 0; t < 8; ++t) { a[i][t] = lo[t]; a[i][8 + t] = hi[t]; }
        }
#pragma unroll
        for (int j = 0; j < 4; ++j)
            b[j] = *(const v16h*)(bptr[j] + kt);
#pragma unroll
        for (int i = 0; i < 4; ++i)
#pragma unroll
            for (int j = 0; j < 4; ++j)
                c[i][j] = __builtin_amdgcn_wmma_f32_16x16x32_f16(
                    false, a[i], false, b[j], (short)0, c[i][j], false, false);
    }
#endif

#pragma unroll
    for (int i = 0; i < 4; ++i)
#pragma unroll
        for (int j = 0; j < 4; ++j) {
            int col = wn + 16 * j + lr;
            int rowBase = wm + 16 * i + ((lane & 16) ? 8 : 0);
            if (col >= N) continue;
#pragma unroll
            for (int r = 0; r < 8; ++r) {
                int row = rowBase + r;
                if (row < M) {
                    long long idx = (long long)row * N + col;
                    float v = c[i][j][r];
                    if (acc) v += C[idx];
                    C[idx] = v;
                }
            }
        }
}

// ---------------- causal depthwise conv (DC=4) + SiLU ----------------
__global__ void conv_silu(const float* __restrict__ xz,
                          const float* __restrict__ cw,
                          const float* __restrict__ cb,
                          _Float16* __restrict__ u16,
                          float* __restrict__ u32) {
    int tid = blockIdx.x * blockDim.x + threadIdx.x;
    if (tid >= TOK * DI_) return;
    int t = tid / DI_, d = tid % DI_;
    int b = t / L_, l = t % L_;
    float acc = cb[d];
#pragma unroll
    for (int j = 0; j < DC_; ++j) {
        int lj = l - (DC_ - 1) + j;
        if (lj >= 0)
            acc += cw[d * DC_ + j] * xz[((long long)(b * L_ + lj)) * (2 * DI_) + d];
    }
    float s = acc / (1.f + __expf(-acc));
    u16[tid] = (_Float16)s;
    u32[tid] = s;
}

// ---------------- dt = softplus(dtgemm + bias) ----------------
__global__ void softplus_bias(float* __restrict__ dt, const float* __restrict__ bias) {
    int tid = blockIdx.x * blockDim.x + threadIdx.x;
    if (tid >= TOK * DI_) return;
    float v = dt[tid] + bias[tid % DI_];
    dt[tid] = (v > 20.f) ? v : log1pf(__expf(v));
}

// ---------------- selective scan (sequential over L) + SiLU(z) gate -> f16 ---
__global__ void selective_scan(const float* __restrict__ dt,
                               const float* __restrict__ u,
                               const float* __restrict__ xz,
                               const float* __restrict__ xdbl,
                               const float* __restrict__ A_log,
                               const float* __restrict__ Dp,
                               _Float16* __restrict__ yg) {
    int tid = blockIdx.x * blockDim.x + threadIdx.x;
    if (tid >= B_ * DI_) return;
    int b = tid / DI_, d = tid % DI_;
    float Arow[DS_], h[DS_];
#pragma unroll
    for (int s = 0; s < DS_; ++s) { Arow[s] = -__expf(A_log[d * DS_ + s]); h[s] = 0.f; }
    float Dd = Dp[d];
    for (int l = 0; l < L_; ++l) {
        long long t = (long long)(b * L_ + l);
        float dtv = dt[t * DI_ + d];
        float uv  = u[t * DI_ + d];
        float zv  = xz[t * (2 * DI_) + DI_ + d];
        const float* Bm = xdbl + t * XD_ + DTR_;
        const float* Cm = Bm + DS_;
        float y = 0.f;
#pragma unroll
        for (int s = 0; s < DS_; ++s) {
            float da = __expf(dtv * Arow[s]);
            h[s] = da * h[s] + dtv * Bm[s] * uv;
            y += h[s] * Cm[s];
        }
        y += Dd * uv;
        float g = zv / (1.f + __expf(-zv));
        yg[t * DI_ + d] = (_Float16)(y * g);
    }
}

// ---------------- host-side orchestration ----------------
static inline int cdiv(long long a, long long b) { return (int)((a + b - 1) / b); }

extern "C" void kernel_launch(void* const* d_in, const int* in_sizes, int n_in,
                              void* d_out, int out_size, void* d_ws, size_t ws_size,
                              hipStream_t stream) {
    (void)in_sizes; (void)n_in; (void)out_size; (void)ws_size;
    const int*   ids       = (const int*)  d_in[0];
    const float* embed     = (const float*)d_in[1];
    const float* norm_w    = (const float*)d_in[2];
    const float* in_proj_w = (const float*)d_in[3];
    const float* conv_w    = (const float*)d_in[4];
    const float* conv_b    = (const float*)d_in[5];
    const float* x_proj_w  = (const float*)d_in[6];
    const float* dt_proj_w = (const float*)d_in[7];
    const float* dt_proj_b = (const float*)d_in[8];
    const float* A_log     = (const float*)d_in[9];
    const float* Dvec      = (const float*)d_in[10];
    const float* out_proj_w= (const float*)d_in[11];
    const float* norm_f_w  = (const float*)d_in[12];
    float* logits = (float*)d_out;

    // workspace carve-up (256B aligned)
    char* ws = (char*)d_ws;
    size_t off = 0;
    auto carve = [&](size_t bytes) -> char* {
        char* p = ws + off;
        off += (bytes + 255) & ~(size_t)255;
        return p;
    };
    float*    x_res   = (float*)   carve((size_t)TOK * DM_ * 4);
    _Float16* h16     = (_Float16*)carve((size_t)TOK * DM_ * 2);
    float*    xz      = (float*)   carve((size_t)TOK * 2 * DI_ * 4);
    _Float16* u16     = (_Float16*)carve((size_t)TOK * DI_ * 2);
    float*    u32     = (float*)   carve((size_t)TOK * DI_ * 4);
    float*    xdbl    = (float*)   carve((size_t)TOK * XD_ * 4);
    _Float16* dtr16   = (_Float16*)carve((size_t)TOK * KDT * 2);
    float*    dtbuf   = (float*)   carve((size_t)TOK * DI_ * 4);
    _Float16* yg16    = (_Float16*)carve((size_t)TOK * DI_ * 2);
    _Float16* w_in16  = (_Float16*)carve((size_t)2 * DI_ * DM_ * 2);
    _Float16* w_x16   = (_Float16*)carve((size_t)XD_ * DI_ * 2);
    _Float16* w_dt16  = (_Float16*)carve((size_t)DI_ * KDT * 2);
    _Float16* w_out16 = (_Float16*)carve((size_t)DM_ * DI_ * 2);
    _Float16* embed16 = (_Float16*)carve((size_t)V_ * DM_ * 2);

    const int TPB = 256;
    auto gemm = [&](const _Float16* A, const _Float16* W, float* C,
                    int M, int N, int K, int acc) {
        dim3 g(cdiv(N, 256), cdiv(M, 128));
        wmma_gemm_tn<<<g, TPB, 0, stream>>>(A, W, C, M, N, K, acc);
    };
    auto cvt = [&](const float* src, int srcLd, int rows, int copyCols,
                   _Float16* dst, int dstLd) {
        long long n = (long long)rows * dstLd;
        convert_pad_f16<<<cdiv(n, TPB), TPB, 0, stream>>>(src, srcLd, rows, copyCols, dst, dstLd);
    };

    // tied LM-head weights -> f16 (once per call; L2-resident thereafter)
    cvt(embed, DM_, V_, DM_, embed16, DM_);

    // x = embed[input_ids]
    embed_gather<<<cdiv((long long)TOK * DM_, TPB), TPB, 0, stream>>>(ids, embed, x_res);

    for (int l = 0; l < NL_; ++l) {
        // h = rms(x, norm_w[l]) -> f16
        rmsnorm_f16<<<TOK, TPB, 0, stream>>>(x_res, norm_w + (size_t)l * DM_, h16, DM_);

        // xz = h @ in_proj_w[l]^T   (1024 x 3072, K=768)
        cvt(in_proj_w + (size_t)l * 2 * DI_ * DM_, DM_, 2 * DI_, DM_, w_in16, DM_);
        gemm(h16, w_in16, xz, TOK, 2 * DI_, DM_, 0);

        // u = silu(causal_conv(xz[:, :DI]) + conv_b)
        conv_silu<<<cdiv((long long)TOK * DI_, TPB), TPB, 0, stream>>>(
            xz, conv_w + (size_t)l * DI_ * DC_, conv_b + (size_t)l * DI_, u16, u32);

        // xdbl = u @ x_proj_w[l]^T  (1024 x 80, K=1536)
        cvt(x_proj_w + (size_t)l * XD_ * DI_, DI_, XD_, DI_, w_x16, DI_);
        gemm(u16, w_x16, xdbl, TOK, XD_, DI_, 0);

        // dt = softplus(dtr @ dt_proj_w[l]^T + b)  (K padded 48 -> 64)
        cvt(xdbl, XD_, TOK, DTR_, dtr16, KDT);
        cvt(dt_proj_w + (size_t)l * DI_ * DTR_, DTR_, DI_, DTR_, w_dt16, KDT);
        gemm(dtr16, w_dt16, dtbuf, TOK, DI_, KDT, 0);
        softplus_bias<<<cdiv((long long)TOK * DI_, TPB), TPB, 0, stream>>>(
            dtbuf, dt_proj_b + (size_t)l * DI_);

        // selective scan + D*u + silu(z) gate -> yg (f16)
        selective_scan<<<cdiv(B_ * DI_, TPB), TPB, 0, stream>>>(
            dtbuf, u32, xz, xdbl, A_log + (size_t)l * DI_ * DS_, Dvec + (size_t)l * DI_, yg16);

        // x += yg @ out_proj_w[l]^T  (1024 x 768, K=1536, accumulate)
        cvt(out_proj_w + (size_t)l * DM_ * DI_, DI_, DM_, DI_, w_out16, DI_);
        gemm(yg16, w_out16, x_res, TOK, DM_, DI_, 1);
    }

    // final RMSNorm + tied LM head: logits = h @ embed^T  (1024 x 32000, K=768)
    rmsnorm_f16<<<TOK, TPB, 0, stream>>>(x_res, norm_f_w, h16, DM_);
    gemm(h16, embed16, logits, TOK, V_, DM_, 0);
}
